// AttentiveDilatedLSTM_3298534883746
// MI455X (gfx1250) — compile-verified, matrix operand order checked by reading, not run on hardware
//
#include <hip/hip_runtime.h>
#include <hip/hip_bf16.h>
#include <cstddef>

#define BB 8
#define TT 128
#define II 256
#define HH 1024
#define AA 256
#define LL 4

typedef __attribute__((ext_vector_type(16))) __bf16 v16bf;
typedef __attribute__((ext_vector_type(8)))  float  v8f;

union Frag { v16bf v; unsigned u[8]; uint4 q[2]; };

__device__ __forceinline__ unsigned short f2bf(float f) {
  unsigned u = __float_as_uint(f);
  u += 0x7FFFu + ((u >> 16) & 1u);            // round-to-nearest-even
  return (unsigned short)(u >> 16);
}
__device__ __forceinline__ float sigmoidf(float x) { return 1.0f / (1.0f + __expf(-x)); }

// 16-bit WMMA A/B lane->K mapping (ISA 7.12.2): lanes 0-15 hold K pairs
// {0..7,16..23}, lanes 16-31 hold {8..15,24..31}; slot vg holds pair (2 elems).
__device__ __forceinline__ int kmap(int lane, int vg) {
  int kb = (lane < 16) ? 0 : 8;
  return (vg < 4) ? (kb + 2 * vg) : (16 + kb + 2 * (vg - 4));
}

// ---------------- workspace layout (bytes) ----------------
static constexpr size_t OFF_WPACK  = 0;           // 4l*256nt*64kt*32lane*8vg uints = 67,108,864 B
static constexpr size_t OFF_WPPACK = 67108864;    // 64nt*8kt*32lane*8vg uints = 524,288 B
static constexpr size_t OFF_XPACK  = 67633152;    // 64mt*8kt*32lane*8vg uints = 524,288 B
static constexpr size_t OFF_LI     = 68157440;    // [B*T][H] f32 = 4,194,304 B
static constexpr size_t OFF_H      = 72351744;    // [L][B][H] f32 = 131,072 B
static constexpr size_t OFF_C      = 72482816;    // [L][B][H] f32
static constexpr size_t OFF_LICUR  = 72613888;    // [B][H] f32
static constexpr size_t OFF_LIWORK = 72646656;    // [B][H] f32
static constexpr size_t OFF_GPART  = 72679424;    // [4ks][4g][64jt][8b][16n] f32 = 524,288 B
static constexpr size_t OFF_HIST   = 73203712;    // sum(Ml)*B*H f32 = 7,864,320 B
static constexpr size_t OFF_SCORES = 81068032;    // [L][B][128] f32 = 16,384 B
static constexpr size_t OUT_BASE   = (size_t)BB * TT * HH; // outputs, then hs, then cs

// ---------------- init: zero h,c ----------------
__global__ void k_init(float* h, float* c) {
  int i = blockIdx.x * 256 + threadIdx.x;
  for (; i < LL * BB * HH; i += 64 * 256) { h[i] = 0.0f; c[i] = 0.0f; }
}

// ---------------- convert + pack (Wih|Whh, Wp, x) to bf16 WMMA fragment layout ----------------
// uint index = (((tile)*32 + lane)*8 + vg); per-lane 8 uints contiguous -> b128 pair loads.
__global__ void k_convert(const float* __restrict__ Wih, const float* __restrict__ Whh,
                          const float* __restrict__ Wp, const float* __restrict__ x,
                          unsigned* __restrict__ wpack, unsigned* __restrict__ wppack,
                          unsigned* __restrict__ xpack) {
  long long idx = (long long)blockIdx.x * 256 + threadIdx.x;
  if (idx < 16777216LL) {                     // big recurrent weights
    long long r = idx;
    int vg   = r & 7;  r >>= 3;
    int lane = r & 31; r >>= 5;
    int kt   = r & 63; r >>= 6;
    int nt   = r & 255; r >>= 8;
    int l = (int)r;
    int n  = (nt << 4) + (lane & 15);
    int k0 = (kt << 5) + kmap(lane, vg);
    float f0, f1;
    if (k0 < HH) {
      const float* p = Wih + ((size_t)(l * 4 * HH + n)) * HH + k0;
      f0 = p[0]; f1 = p[1];
    } else {
      const float* p = Whh + ((size_t)(l * 4 * HH + n)) * HH + (k0 - HH);
      f0 = p[0]; f1 = p[1];
    }
    wpack[idx] = (unsigned)f2bf(f0) | ((unsigned)f2bf(f1) << 16);
  } else if (idx < 16777216LL + 131072LL) {   // Wp (B operand of input proj)
    long long j = idx - 16777216LL;
    int vg = j & 7, lane = (int)((j >> 3) & 31), kt = (int)((j >> 8) & 7), nt = (int)(j >> 11);
    int n  = nt * 16 + (lane & 15);
    int k0 = kt * 32 + kmap(lane, vg);
    wppack[j] = (unsigned)f2bf(Wp[(size_t)n * II + k0]) |
                ((unsigned)f2bf(Wp[(size_t)n * II + k0 + 1]) << 16);
  } else if (idx < 16777216LL + 262144LL) {   // x (A operand of input proj)
    long long j = idx - 16777216LL - 131072LL;
    int vg = j & 7, lane = (int)((j >> 3) & 31), kt = (int)((j >> 8) & 7), mt = (int)(j >> 11);
    int m  = mt * 16 + (lane & 15);
    int k0 = kt * 32 + kmap(lane, vg);
    xpack[j] = (unsigned)f2bf(x[(size_t)m * II + k0]) |
               ((unsigned)f2bf(x[(size_t)m * II + k0 + 1]) << 16);
  }
}

// ---------------- input projection: Li[b*T+t][n] = x @ Wp^T + bp (WMMA bf16) ----------------
__global__ void k_inproj(const unsigned* __restrict__ xpack, const float* __restrict__ bp,
                         const unsigned* __restrict__ wppack, float* __restrict__ Li) {
  int lane = threadIdx.x;
  int m0 = blockIdx.x * 16, n0 = blockIdx.y * 16;
  const unsigned* abase = xpack + ((size_t)(blockIdx.x * 8) * 32 + lane) * 8;
  const unsigned* bbase = wppack + ((size_t)(blockIdx.y * 8) * 32 + lane) * 8;
  v8f acc0 = {}, acc1 = {};
#pragma unroll
  for (int kp = 0; kp < 4; ++kp) {            // two independent accumulator chains
    Frag a0, b0, a1, b1;
    const uint4* a0q = (const uint4*)(abase + (2 * kp) * 256);
    const uint4* b0q = (const uint4*)(bbase + (2 * kp) * 256);
    const uint4* a1q = (const uint4*)(abase + (2 * kp + 1) * 256);
    const uint4* b1q = (const uint4*)(bbase + (2 * kp + 1) * 256);
    a0.q[0] = a0q[0]; a0.q[1] = a0q[1];
    b0.q[0] = b0q[0]; b0.q[1] = b0q[1];
    a1.q[0] = a1q[0]; a1.q[1] = a1q[1];
    b1.q[0] = b1q[0]; b1.q[1] = b1q[1];
    acc0 = __builtin_amdgcn_wmma_f32_16x16x32_bf16(false, a0.v, false, b0.v, (short)0, acc0,
                                                   false, false);
    acc1 = __builtin_amdgcn_wmma_f32_16x16x32_bf16(false, a1.v, false, b1.v, (short)0, acc1,
                                                   false, false);
  }
  v8f acc = acc0 + acc1;
  float bias = bp[n0 + (lane & 15)];
#pragma unroll
  for (int r = 0; r < 8; ++r) {
    int M = m0 + r + ((lane < 16) ? 0 : 8);
    Li[(size_t)M * HH + n0 + (lane & 15)] = acc[r] + bias;
  }
}

// ---------------- attention: append h to hist, score the new entry ----------------
__global__ void k_attn_score(const float* __restrict__ Wa, const float* __restrict__ ba,
                             const float* __restrict__ wc, const float* __restrict__ bc,
                             const float* __restrict__ h, float* __restrict__ hist,
                             float* __restrict__ scores, int l, int cnt, int Ml) {
  int b = blockIdx.x, a = threadIdx.x;
  const float* hb = h + ((size_t)(l * BB + b)) * HH;
  float* dst = hist + ((size_t)(b * Ml + (cnt - 1))) * HH;
  for (int j = a; j < HH; j += 256) dst[j] = hb[j];   // pre-update h appended
  const float* wrow = Wa + ((size_t)(l * AA + a)) * HH;
  float s = 0.0f;
  for (int j = 0; j < HH; ++j) s += wrow[j] * hb[j];
  float v = tanhf(s + ba[l * AA + a]) * wc[l * AA + a];
  __shared__ float red[256];
  red[a] = v;
  __syncthreads();
  for (int st = 128; st > 0; st >>= 1) {
    if (a < st) red[a] += red[a + st];
    __syncthreads();
  }
  if (a == 0) scores[(l * BB + b) * 128 + (cnt - 1)] = red[0] + bc[l];
}

// ---------------- softmax over history + context; liwork = li (+ ctx) ----------------
__global__ void k_attn_ctx(const float* __restrict__ hist, const float* __restrict__ scores,
                           const float* __restrict__ Li, const float* __restrict__ licur,
                           float* __restrict__ liwork, int l, int t, int cnt, int useCtx,
                           int Ml) {
  __shared__ float w[BB * 128];
  __shared__ float mx[BB], inv[BB];
  int tid = threadIdx.x;
  if (useCtx) {
    for (int i = tid; i < BB * cnt; i += 256) {
      int b = i / cnt, m = i - b * cnt;
      w[b * 128 + m] = scores[(l * BB + b) * 128 + m];
    }
    __syncthreads();
    if (tid < BB) {
      float m = -1e30f;
      for (int j = 0; j < cnt; ++j) m = fmaxf(m, w[tid * 128 + j]);
      float s = 0.0f;
      for (int j = 0; j < cnt; ++j) s += __expf(w[tid * 128 + j] - m);
      mx[tid] = m; inv[tid] = 1.0f / s;
    }
    __syncthreads();
    for (int i = tid; i < BB * cnt; i += 256) {
      int b = i / cnt, m = i - b * cnt;
      w[b * 128 + m] = __expf(w[b * 128 + m] - mx[b]) * inv[b];
    }
    __syncthreads();
  }
  for (int e = blockIdx.x * 256 + tid; e < BB * HH; e += 16 * 256) {
    int b = e >> 10, j = e & 1023;
    float li = (l == 0) ? Li[((size_t)(b * TT + t)) * HH + j] : licur[b * HH + j];
    if (useCtx) {
      float cx = 0.0f;
      const float* hp = hist + ((size_t)b * Ml) * HH + j;
      for (int m = 0; m < cnt; ++m) cx += w[b * 128 + m] * hp[(size_t)m * HH];
      li += cx;
    }
    liwork[b * HH + j] = li;
  }
}

// ---------------- LSTM gates (WMMA bf16, K-split partials) ----------------
// grid (64 j-tiles, 4 K-chunks), 4 waves/block (gate i/f/g/o); K-chunk = 512
__global__ void k_gates(const unsigned* __restrict__ wpackL, const float* __restrict__ liwork,
                        const float* __restrict__ hist, float* __restrict__ gpart, int cnt,
                        int Ml) {
  __shared__ unsigned a_lds[8 * 257];           // 8 rows x 512 bf16 (K-chunk), bank-padded
  int tid = threadIdx.x;
  int jt = blockIdx.x, ks = blockIdx.y;
  for (int i = tid; i < 8 * 256; i += 128) {
    int b = i >> 8;
    int kk = ks * 512 + (i & 255) * 2;
    float f0, f1;
    if (kk < HH) {
      f0 = liwork[b * HH + kk]; f1 = liwork[b * HH + kk + 1];
    } else {
      const float* hp = hist + ((size_t)(b * Ml + (cnt - 1))) * HH + (kk - HH);
      f0 = hp[0]; f1 = hp[1];
    }
    a_lds[b * 257 + (i & 255)] = (unsigned)f2bf(f0) | ((unsigned)f2bf(f1) << 16);
  }
  __syncthreads();

  int wv = tid >> 5, lane = tid & 31;
  int ntile = wv * 64 + jt;                     // W row tile (gate wv, cols j0..j0+15)
  int m = lane & 15;
  bool mv = m < 8;
  const unsigned* bbase = wpackL + ((size_t)(ntile * 64 + ks * 16) * 32 + lane) * 8;
  const unsigned* albase = a_lds + m * 257;
  v8f acc0 = {}, acc1 = {};
#pragma unroll
  for (int kp = 0; kp < 8; ++kp) {              // two independent accumulator chains
    Frag a0, b0, a1, b1;
#pragma unroll
    for (int vg = 0; vg < 8; ++vg) {
      int km = kmap(lane, vg) >> 1;
      a0.u[vg] = mv ? albase[(2 * kp) * 16 + km] : 0u;
      a1.u[vg] = mv ? albase[(2 * kp + 1) * 16 + km] : 0u;
    }
    const uint4* b0q = (const uint4*)(bbase + (2 * kp) * 256);
    const uint4* b1q = (const uint4*)(bbase + (2 * kp + 1) * 256);
    b0.q[0] = b0q[0]; b0.q[1] = b0q[1];
    b1.q[0] = b1q[0]; b1.q[1] = b1q[1];
    __builtin_prefetch((const void*)(bbase + (2 * kp + 2) * 256), 0, 0);  // next K-tiles
    acc0 = __builtin_amdgcn_wmma_f32_16x16x32_bf16(false, a0.v, false, b0.v, (short)0, acc0,
                                                   false, false);
    acc1 = __builtin_amdgcn_wmma_f32_16x16x32_bf16(false, a1.v, false, b1.v, (short)0, acc1,
                                                   false, false);
  }
  v8f acc = acc0 + acc1;
  if (lane < 16) {                              // C rows M=r live in lanes 0-15
#pragma unroll
    for (int r = 0; r < 8; ++r)
      gpart[((size_t)(((ks * 4 + wv) * 64 + jt) * 8 + r)) * 16 + lane] = acc[r];
  }
}

// ---------------- reduce K-partials + LSTM cell activations ----------------
__global__ void k_cell(const float* __restrict__ gpart, const float* __restrict__ bih,
                       const float* __restrict__ bhh, float* __restrict__ h,
                       float* __restrict__ c, float* __restrict__ licur, int l) {
  int e = blockIdx.x * 256 + threadIdx.x;       // 8192 threads: (b, j)
  int b = e >> 10, j = e & 1023;
  int jt = j >> 4, jj = j & 15;
  float g4[4];
#pragma unroll
  for (int g = 0; g < 4; ++g) {
    float s = 0.0f;
#pragma unroll
    for (int ks = 0; ks < 4; ++ks)
      s += gpart[((size_t)(((ks * 4 + g) * 64 + jt) * 8 + b)) * 16 + jj];
    g4[g] = s + bih[l * 4 * HH + g * HH + j] + bhh[l * 4 * HH + g * HH + j];
  }
  size_t si = ((size_t)(l * BB + b)) * HH + j;
  float cn = sigmoidf(g4[1]) * c[si] + sigmoidf(g4[0]) * tanhf(g4[2]);
  float hn = sigmoidf(g4[3]) * tanhf(cn);
  c[si] = cn; h[si] = hn; licur[b * HH + j] = hn;
}

// ---------------- emit outputs for t..t+7 (h3 constant between dilation-8 updates) ----------------
__global__ void k_copyout(const float* __restrict__ h, float* __restrict__ out, int t) {
  for (int e = blockIdx.x * 256 + threadIdx.x; e < BB * 8 * HH; e += 32 * 256) {
    int j = e & 1023, dt = (e >> 10) & 7, b = e >> 13;
    out[((size_t)(b * TT + t + dt)) * HH + j] = h[((size_t)(3 * BB + b)) * HH + j];
  }
}

// ---------------- final (hs, cs) ----------------
__global__ void k_final(const float* __restrict__ h, const float* __restrict__ c,
                        float* __restrict__ out) {
  int i = blockIdx.x * 256 + threadIdx.x;
  for (; i < 2 * LL * BB * HH; i += 64 * 256)
    out[OUT_BASE + i] = (i < LL * BB * HH) ? h[i] : c[i - LL * BB * HH];
}

extern "C" void kernel_launch(void* const* d_in, const int* in_sizes, int n_in, void* d_out,
                              int out_size, void* d_ws, size_t ws_size, hipStream_t stream) {
  const float* x   = (const float*)d_in[0];
  const float* Wp  = (const float*)d_in[1];
  const float* bp  = (const float*)d_in[2];
  const float* Wih = (const float*)d_in[3];
  const float* bih = (const float*)d_in[4];
  const float* Whh = (const float*)d_in[5];
  const float* bhh = (const float*)d_in[6];
  const float* Wa  = (const float*)d_in[7];
  const float* ba  = (const float*)d_in[8];
  const float* wc  = (const float*)d_in[9];
  const float* bc  = (const float*)d_in[10];
  float* out = (float*)d_out;

  char* ws = (char*)d_ws;
  unsigned* wpack  = (unsigned*)(ws + OFF_WPACK);
  unsigned* wppack = (unsigned*)(ws + OFF_WPPACK);
  unsigned* xpack  = (unsigned*)(ws + OFF_XPACK);
  float* Li     = (float*)(ws + OFF_LI);
  float* h      = (float*)(ws + OFF_H);
  float* c      = (float*)(ws + OFF_C);
  float* licur  = (float*)(ws + OFF_LICUR);
  float* liwork = (float*)(ws + OFF_LIWORK);
  float* gpart  = (float*)(ws + OFF_GPART);
  float* hist0  = (float*)(ws + OFF_HIST);
  float* scores = (float*)(ws + OFF_SCORES);

  static const int dil[LL]     = {1, 2, 4, 8};
  static const int Ml[LL]      = {128, 64, 32, 16};
  static const int histcum[LL] = {0, 128, 192, 224};

  k_init<<<64, 256, 0, stream>>>(h, c);
  k_convert<<<(16777216 + 262144 + 255) / 256, 256, 0, stream>>>(Wih, Whh, Wp, x, wpack,
                                                                 wppack, xpack);
  k_inproj<<<dim3(64, 64), 32, 0, stream>>>(xpack, bp, wppack, Li);

  for (int t = 0; t < TT; ++t) {
    for (int l = 0; l < LL; ++l) {
      if (t % dil[l] != 0) continue;
      int cnt = t / dil[l] + 1;
      float* histL = hist0 + (size_t)histcum[l] * BB * HH;
      const unsigned* wpackL = wpack + (size_t)l * 256 * 64 * 8 * 32;
      k_attn_score<<<8, 256, 0, stream>>>(Wa, ba, wc, bc, h, histL, scores, l, cnt, Ml[l]);
      k_attn_ctx<<<16, 256, 0, stream>>>(histL, scores, Li, licur, liwork, l, t, cnt,
                                         (cnt > 1) ? 1 : 0, Ml[l]);
      k_gates<<<dim3(64, 4), 128, 0, stream>>>(wpackL, liwork, histL, gpart, cnt, Ml[l]);
      k_cell<<<32, 256, 0, stream>>>(gpart, bih, bhh, h, c, licur, l);
    }
    if ((t & 7) == 0) k_copyout<<<32, 256, 0, stream>>>(h, out, t);
  }
  k_final<<<64, 256, 0, stream>>>(h, c, out);
}
